// SAKT_12068858102382
// MI455X (gfx1250) — matrix operand order repaired and checked by reading, not compiled, not run on hardware
//
#include <hip/hip_runtime.h>

// ---------------------------------------------------------------------------
// SAKT (knowledge-tracing transformer) forward for MI455X / gfx1250.
// B=8 S=1024 E=512 H=8 D=64 L=2 F=2048. All GEMM-class math runs through
// v_wmma_f32_16x16x32_f16 (f16 in / f32 accumulate). Relative-position terms
// use the rank-64 structure of rel=clip(i-j,0,63): qp-precompute + gather for
// the key side, 64-bucket row compression + tiny WMMA for the value side, so
// the 256MB [S,S,D] pos tensors are never materialized.
// New this round: GEMM weight tiles are staged into LDS by the Tensor Data
// Mover (tensor_load_to_lds + s_wait_tensorcnt), B-fragments are fed from LDS,
// and the streaming A operand is prefetched (global_prefetch_b8).
// Workspace use: ~156 MB.
// ---------------------------------------------------------------------------

typedef __attribute__((ext_vector_type(16))) _Float16 v16h;
typedef __attribute__((ext_vector_type(8)))  float    v8f;
typedef __attribute__((ext_vector_type(4)))  unsigned int u32x4;
typedef __attribute__((ext_vector_type(8)))  int      i32x8;
typedef __attribute__((ext_vector_type(4)))  int      i32x4;

#define SEQ   1024
#define EMB   512
#define NH    8
#define HD    64
#define FF    2048
#define ROWS  8192          // B*S
#define SCLD  1064          // score-row leading dim (1024 + 32 pad + 8)

// ---------------- TDM: DMA a [rows x rowlen] f16 tile (row stride `stride`
// elements) from global memory into LDS. Issued by one wave; caller barriers.
// D# layout per CDNA5 ISA 8.3/8.4 (group0: count/lds/global/type,
// group1: data_size, tensor dims, tile dims, dim0 stride).
__device__ inline void tdm_load_2d_f16(const _Float16* gsrc, _Float16* ldsdst,
                                       int rows, int rowlen, int stride) {
#if __has_builtin(__builtin_amdgcn_tensor_load_to_lds)
  unsigned long long ga = (unsigned long long)(size_t)gsrc;
  unsigned lds_addr = (unsigned)(size_t)ldsdst;     // flat low 32 bits == LDS addr
  u32x4 g0;
  g0[0] = 1u;                                        // count=1, user descriptor
  g0[1] = lds_addr;                                  // LDS byte address
  g0[2] = (unsigned)(ga & 0xffffffffu);              // global_addr[31:0]
  g0[3] = (unsigned)((ga >> 32) & 0x01ffffffu)       // global_addr[56:32]
          | 0x80000000u;                             // type=2 ("image")
  i32x8 g1;
  g1[0] = 1 << 16;                                   // data_size=1 (2B), no mcast
  g1[1] = (rowlen & 0xffff) << 16;                   // tensor_dim0[15:0]
  g1[2] = ((rowlen >> 16) & 0xffff) | ((rows & 0xffff) << 16); // d0 hi | d1 lo
  g1[3] = ((rows >> 16) & 0xffff) | ((rowlen & 0xffff) << 16); // d1 hi | tile_dim0
  g1[4] = rows & 0xffff;                             // tile_dim1 | tile_dim2=0
  g1[5] = stride;                                    // tensor_dim0_stride[31:0]
  g1[6] = 0;                                         // stride hi | dim1_stride lo
  g1[7] = 0;
  i32x4 z4 = {0, 0, 0, 0};
#if defined(__clang_major__) && (__clang_major__ >= 23)
  i32x8 z8 = {0, 0, 0, 0, 0, 0, 0, 0};
  __builtin_amdgcn_tensor_load_to_lds(g0, g1, z4, z4, z8, 0);
#else
  __builtin_amdgcn_tensor_load_to_lds(g0, g1, z4, z4, 0);
#endif
  __builtin_amdgcn_s_wait_tensorcnt(0);
#else
  // Fallback: cooperative copy by the issuing wave.
  const int lane = threadIdx.x & 31;
  for (int t = lane; t < rows * rowlen; t += 32) {
    int r = t / rowlen, c = t - r * rowlen;
    ldsdst[t] = gsrc[(size_t)r * stride + c];
  }
#endif
}

// ---------------- weight prep: f32 -> f16, optionally transposed -----------
__global__ void f32_to_f16_T(const float* __restrict__ src,
                             _Float16* __restrict__ dst, int R, int C) {
  int idx = blockIdx.x * 256 + threadIdx.x;
  if (idx >= R * C) return;
  int r = idx / C, c = idx - r * C;
  dst[(size_t)c * R + r] = (_Float16)src[idx];   // dst is [C][R] (B^T, row-major)
}

__global__ void f32_to_f16(const float* __restrict__ src,
                           _Float16* __restrict__ dst, int n) {
  int idx = blockIdx.x * 256 + threadIdx.x;
  if (idx < n) dst[idx] = (_Float16)src[idx];
}

// ---------------- embedding gather + label gating --------------------------
__global__ __launch_bounds__(256) void embed_k(
    const int* __restrict__ it_in, const int* __restrict__ sk_in,
    const int* __restrict__ lbl, const int* __restrict__ it_id,
    const int* __restrict__ sk_id,
    const float* __restrict__ iemb, const float* __restrict__ semb,
    _Float16* __restrict__ x2e, _Float16* __restrict__ qh)
{
  const int row = blockIdx.x;          // b*S + s
  const int c = threadIdx.x;           // 0..255  (E/2)
  const float l = (float)lbl[row];
  const float ie = iemb[(size_t)it_in[row] * 256 + c];
  const float se = semb[(size_t)sk_in[row] * 256 + c];
  _Float16* xr = x2e + (size_t)row * 1024;
  xr[c]       = (_Float16)(ie * l);
  xr[256 + c] = (_Float16)(se * l);
  xr[512 + c] = (_Float16)(ie * (1.f - l));
  xr[768 + c] = (_Float16)(se * (1.f - l));
  _Float16* qr = qh + (size_t)row * EMB;
  qr[c]       = (_Float16)iemb[(size_t)it_id[row] * 256 + c];
  qr[256 + c] = (_Float16)semb[(size_t)sk_id[row] * 256 + c];
}

// ---------------- generic WMMA GEMM:  C = act(A @ B + bias) ----------------
// A: [M][K] f16 row-major (lda). BT: [N][K] f16 row-major (ldb) = B^T.
// 4 waves/block, each wave owns one 16x16 C tile; grid = (N/16, M/64).
// The block's 16xK B^T tile is DMA'd into LDS once by the TDM, then all four
// waves feed their B-fragments from LDS while streaming (+prefetching) A.
// Dynamic LDS: 16*K*2 bytes (<=64KB for K=2048; WGP has 320KB).
__global__ __launch_bounds__(128) void gemm_wmma_k(
    const _Float16* __restrict__ A, int lda,
    const _Float16* __restrict__ BT, int ldb,
    const float* __restrict__ bias,
    float* __restrict__ Cf, _Float16* __restrict__ Ch, int ldc,
    int K, int relu)
{
  extern __shared__ _Float16 sB[];      // [16][K]
  const int lane = threadIdx.x & 31;
  const int wave = threadIdx.x >> 5;
  const int tn = blockIdx.x * 16;
  const int tm = blockIdx.y * 64 + wave * 16;
  const int l15 = lane & 15;
  const int ka  = (lane >> 4) << 3;     // A-frag K base, also C-row base
  const int kbb = (lane >> 4) << 4;     // unused (B-frag K base is ka)

  if (wave == 0)
    tdm_load_2d_f16(BT + (size_t)tn * ldb, sB, 16, K, ldb);
  __syncthreads();

  const _Float16* arow = A  + (size_t)(tm + l15) * lda;
  const _Float16* brow = sB + (size_t)l15 * K;     // LDS row for this lane's N
  v8f acc = {};
  for (int k0 = 0; k0 < K; k0 += 32) {
    if (k0 + 256 < K) __builtin_prefetch(arow + k0 + 256, 0, 1);
    v16h a, b;
#pragma unroll
    for (int t = 0; t < 8; ++t) {
      a[t]     = arow[k0 + ka + t];
      a[8 + t] = arow[k0 + ka + 16 + t];
    }
#pragma unroll
    for (int t = 0; t < 16; ++t) b[t] = brow[k0 + ka + (t & 7) + ((t >> 3) << 4)];
    acc = __builtin_amdgcn_wmma_f32_16x16x32_f16(false, a, false, b,
                                                 (short)0, acc, false, false);
  }
  const int cn = tn + l15;
  const float bvv = bias ? bias[cn] : 0.f;
#pragma unroll
  for (int r = 0; r < 8; ++r) {
    const int cm = tm + r + ka;
    float v = acc[r] + bvv;
    if (relu) v = fmaxf(v, 0.f);
    if (Cf) Cf[(size_t)cm * ldc + cn] = v;
    if (Ch) Ch[(size_t)cm * ldc + cn] = (_Float16)v;
  }
}

// ---------------- fused causal attention with relative positions -----------
// One block = 16 query rows of one (b,h). 128 threads (4 waves).
// Dynamic LDS: sc[16][SCLD] f32 scores/probs + pw[16][64] f32 buckets (~71KB
// of the 320KB WGP LDS). Phase1: QK^T via WMMA + qp gather + mask -> LDS.
// Phase2: softmax + deterministic 64-bucket compression of each prob row.
// Phase3: out = P@V (WMMA, A-frags fed from LDS) + PW@posV (WMMA).
__global__ __launch_bounds__(128) void attn_k(
    const _Float16* __restrict__ qh, const _Float16* __restrict__ kh,
    const _Float16* __restrict__ vh, const float* __restrict__ qp,
    const _Float16* __restrict__ posvT, float* __restrict__ outf)
{
  extern __shared__ char smem[];
  float* sc = (float*)smem;                                  // [16][SCLD]
  float* pw = (float*)(smem + (size_t)16 * SCLD * 4);        // [16][64]
  const int bi = blockIdx.x;            // query tile, 0..63
  const int b  = blockIdx.y >> 3;
  const int h  = blockIdx.y & 7;
  const int i0 = bi * 16;
  const int tid  = threadIdx.x;
  const int lane = tid & 31;
  const int wave = tid >> 5;
  const int l15  = lane & 15;
  const int ka   = (lane >> 4) << 3;
  const int kbb  = (lane >> 4) << 4;
  const int ntile = bi + 1;
  const int ncol  = ntile * 16;

  for (int t = tid; t < 16 * 64; t += 128) pw[t] = 0.f;

  // Q fragments for this 16-row tile (K = head dim 0..63 -> two A frags)
  const _Float16* arow = qh + ((size_t)(b * SEQ + i0 + l15) * EMB + h * HD);
  v16h a0, a1;
#pragma unroll
  for (int t = 0; t < 8; ++t) {
    a0[t] = arow[ka + t];       a0[8 + t] = arow[ka + 16 + t];
    a1[t] = arow[32 + ka + t];  a1[8 + t] = arow[32 + ka + 16 + t];
  }
  // phase 1: scores = (q.k + qp[clip(i-j)]) * scale, causal mask
  for (int jt = wave; jt < ntile; jt += 4) {
    const int j0 = jt * 16;
    const _Float16* brow = kh + ((size_t)(b * SEQ + j0 + l15) * EMB + h * HD);
    v16h b0, b1;
#pragma unroll
    for (int t = 0; t < 16; ++t) { b0[t] = brow[ka + (t & 7) + ((t >> 3) << 4)]; b1[t] = brow[32 + ka + (t & 7) + ((t >> 3) << 4)]; }
    v8f s = {};
    s = __builtin_amdgcn_wmma_f32_16x16x32_f16(false, a0, false, b0, (short)0, s, false, false);
    s = __builtin_amdgcn_wmma_f32_16x16x32_f16(false, a1, false, b1, (short)0, s, false, false);
    const int j = j0 + l15;
#pragma unroll
    for (int r = 0; r < 8; ++r) {
      const int i = i0 + r + ka;
      float v;
      if (j > i) v = -1e9f;
      else {
        int rel = i - j; if (rel > 63) rel = 63;
        v = (s[r] + qp[(size_t)(b * SEQ + i) * EMB + h * HD + rel]) * 0.125f;
      }
      sc[(r + ka) * SCLD + j] = v;
    }
  }
  __syncthreads();

  // phase 2: softmax + bucket compression (deterministic: rel%8 is fixed per
  // sub-thread, so only the clipped bucket 63 needs a reduction)
  {
    const int row = tid >> 3, sub = tid & 7;
    const int i = i0 + row;
    float* srow = sc + row * SCLD;
    float m = -1e30f;
    for (int j = sub; j < ncol; j += 8) m = fmaxf(m, srow[j]);
#pragma unroll
    for (int d = 1; d < 8; d <<= 1) m = fmaxf(m, __shfl_xor(m, d, 32));
    float sum = 0.f;
    for (int j = sub; j < ncol; j += 8) {
      float e = __expf(srow[j] - m); srow[j] = e; sum += e;
    }
#pragma unroll
    for (int d = 1; d < 8; d <<= 1) sum += __shfl_xor(sum, d, 32);
    const float inv = 1.f / sum;
    float c63 = 0.f;
    for (int j = sub; j < ncol; j += 8) {
      float p = srow[j] * inv;
      srow[j] = p;
      if (j <= i) {
        int rel = i - j;
        if (rel >= 63) c63 += p; else pw[row * 64 + rel] += p;
      }
    }
#pragma unroll
    for (int d = 1; d < 8; d <<= 1) c63 += __shfl_xor(c63, d, 32);
    if (sub == 0) pw[row * 64 + 63] += c63;
    for (int j = ncol + sub; j < ncol + 32; j += 8) srow[j] = 0.f;  // K-tail pad
  }
  __syncthreads();

  // phase 3: out = P@V + PW@posV ; this wave owns d-chunk wave*16
  const int d0 = wave * 16;
  v8f acc = {};
  for (int j0 = 0; j0 < ncol; j0 += 32) {
    v16h a, bm;
#pragma unroll
    for (int t = 0; t < 8; ++t) {
      a[t]     = (_Float16)sc[l15 * SCLD + j0 + ka + t];
      a[8 + t] = (_Float16)sc[l15 * SCLD + j0 + ka + 16 + t];
    }
#pragma unroll
    for (int t = 0; t < 16; ++t) {
      int k = j0 + ka + (t & 7) + ((t >> 3) << 4); if (k > SEQ - 1) k = SEQ - 1;   // pad rows * P==0
      bm[t] = vh[(size_t)(b * SEQ + k) * EMB + h * HD + d0 + l15];
    }
    acc = __builtin_amdgcn_wmma_f32_16x16x32_f16(false, a, false, bm,
                                                 (short)0, acc, false, false);
  }
  {
    const _Float16* brow = posvT + (size_t)(d0 + l15) * 64;
#pragma unroll
    for (int k0 = 0; k0 < 64; k0 += 32) {
      v16h a, bm;
#pragma unroll
      for (int t = 0; t < 8; ++t) {
        a[t]     = (_Float16)pw[l15 * 64 + k0 + ka + t];
        a[8 + t] = (_Float16)pw[l15 * 64 + k0 + ka + 16 + t];
      }
#pragma unroll
      for (int t = 0; t < 16; ++t) bm[t] = brow[k0 + ka + (t & 7) + ((t >> 3) << 4)];
      acc = __builtin_amdgcn_wmma_f32_16x16x32_f16(false, a, false, bm,
                                                   (short)0, acc, false, false);
    }
  }
  const int cn = h * HD + d0 + l15;
#pragma unroll
  for (int r = 0; r < 8; ++r) {
    const int i = i0 + r + ka;
    outf[(size_t)(b * SEQ + i) * EMB + cn] = acc[r];
  }
}

// ---------------- residual add + layernorm (fp32), emits f16 copy ----------
__global__ __launch_bounds__(256) void add_ln_k(
    float* __restrict__ x, const float* __restrict__ add,
    const float* __restrict__ w, const float* __restrict__ bb,
    _Float16* __restrict__ xh)
{
  __shared__ float red[8];
  const int row = blockIdx.x;
  const int tid = threadIdx.x;
  const size_t base = (size_t)row * EMB;
  float v0 = x[base + tid] + add[base + tid];
  float v1 = x[base + 256 + tid] + add[base + 256 + tid];
  float s = v0 + v1;
#pragma unroll
  for (int d = 1; d < 32; d <<= 1) s += __shfl_xor(s, d, 32);
  if ((tid & 31) == 0) red[tid >> 5] = s;
  __syncthreads();
  float mu = 0.f;
#pragma unroll
  for (int i = 0; i < 8; ++i) mu += red[i];
  mu *= (1.f / 512.f);
  __syncthreads();
  float d0 = v0 - mu, d1 = v1 - mu;
  s = d0 * d0 + d1 * d1;
#pragma unroll
  for (int d = 1; d < 32; d <<= 1) s += __shfl_xor(s, d, 32);
  if ((tid & 31) == 0) red[tid >> 5] = s;
  __syncthreads();
  float var = 0.f;
#pragma unroll
  for (int i = 0; i < 8; ++i) var += red[i];
  var *= (1.f / 512.f);
  const float rs = rsqrtf(var + 1e-5f);
  float y0 = d0 * rs * w[tid] + bb[tid];
  float y1 = d1 * rs * w[256 + tid] + bb[256 + tid];
  x[base + tid] = y0;        x[base + 256 + tid] = y1;
  xh[base + tid] = (_Float16)y0; xh[base + 256 + tid] = (_Float16)y1;
}

// ---------------- final projection to 1 output per row ---------------------
__global__ __launch_bounds__(256) void final_k(
    const float* __restrict__ x, const float* __restrict__ w,
    const float* __restrict__ b, float* __restrict__ out)
{
  __shared__ float red[8];
  const int row = blockIdx.x, tid = threadIdx.x;
  const size_t base = (size_t)row * EMB;
  float s = x[base + tid] * w[tid] + x[base + 256 + tid] * w[256 + tid];
#pragma unroll
  for (int d = 1; d < 32; d <<= 1) s += __shfl_xor(s, d, 32);
  if ((tid & 31) == 0) red[tid >> 5] = s;
  __syncthreads();
  if (tid == 0) {
    float t = b[0];
#pragma unroll
    for (int i = 0; i < 8; ++i) t += red[i];
    out[row] = t;
  }
}

// ---------------------------------------------------------------------------
extern "C" void kernel_launch(void* const* d_in, const int* in_sizes, int n_in,
                              void* d_out, int out_size, void* d_ws, size_t ws_size,
                              hipStream_t stream) {
  (void)in_sizes; (void)n_in; (void)out_size; (void)ws_size;
  const int*   item_inputs  = (const int*)  d_in[0];
  const int*   skill_inputs = (const int*)  d_in[1];
  const int*   label_inputs = (const int*)  d_in[2];
  const int*   item_ids     = (const int*)  d_in[3];
  const int*   skill_ids    = (const int*)  d_in[4];
  const float* item_emb     = (const float*)d_in[5];
  const float* skill_emb    = (const float*)d_in[6];
  const float* pos_key_emb  = (const float*)d_in[7];
  const float* pos_value_emb= (const float*)d_in[8];
  const float* lin_in_w     = (const float*)d_in[9];
  const float* lin_in_b     = (const float*)d_in[10];
  const float* Wq           = (const float*)d_in[11];
  const float* bq           = (const float*)d_in[12];
  const float* Wk           = (const float*)d_in[13];
  const float* bk           = (const float*)d_in[14];
  const float* Wv           = (const float*)d_in[15];
  const float* bv           = (const float*)d_in[16];
  const float* ln_w         = (const float*)d_in[17];
  const float* ln_b         = (const float*)d_in[18];
  const float* ff1_w        = (const float*)d_in[19];
  const float* ff1_b        = (const float*)d_in[20];
  const float* ff2_w        = (const float*)d_in[21];
  const float* ff2_b        = (const float*)d_in[22];
  const float* lin_out_w    = (const float*)d_in[23];
  const float* lin_out_b    = (const float*)d_in[24];

  // ---- workspace carve-up (~156 MB) ----
  char* base = (char*)d_ws;
  size_t off = 0;
  auto A_ = [&](size_t bytes) -> char* {
    char* p = base + off;
    off += (bytes + 255) & ~(size_t)255;
    return p;
  };
  _Float16* h_lininT = (_Float16*)A_((size_t)512 * 1024 * 2);   // [512][1024]
  _Float16* h_WqT[2], *h_WkT[2], *h_WvT[2], *h_ff1T[2], *h_ff2T[2];
  for (int i = 0; i < 2; ++i) {
    h_WqT[i]  = (_Float16*)A_((size_t)512 * 512 * 2);           // [512][512]
    h_WkT[i]  = (_Float16*)A_((size_t)512 * 512 * 2);
    h_WvT[i]  = (_Float16*)A_((size_t)512 * 512 * 2);
    h_ff1T[i] = (_Float16*)A_((size_t)2048 * 512 * 2);          // [2048][512]
    h_ff2T[i] = (_Float16*)A_((size_t)512 * 2048 * 2);          // [512][2048]
  }
  _Float16* h_posk  = (_Float16*)A_((size_t)64 * 64 * 2);       // [r][d]
  _Float16* h_posvT = (_Float16*)A_((size_t)64 * 64 * 2);       // [d][r]
  _Float16* x2e_h   = (_Float16*)A_((size_t)ROWS * 1024 * 2);
  _Float16* query_h = (_Float16*)A_((size_t)ROWS * EMB * 2);
  float*    inputs_f= (float*)   A_((size_t)ROWS * EMB * 4);
  _Float16* inputs_h= (_Float16*)A_((size_t)ROWS * EMB * 2);
  _Float16* q_h     = (_Float16*)A_((size_t)ROWS * EMB * 2);
  _Float16* k_h     = (_Float16*)A_((size_t)ROWS * EMB * 2);
  _Float16* v_h     = (_Float16*)A_((size_t)ROWS * EMB * 2);
  float*    qp_f    = (float*)   A_((size_t)ROWS * EMB * 4);    // [row][h*64+rel]
  float*    attn_f  = (float*)   A_((size_t)ROWS * EMB * 4);    // reused for FF2 out
  _Float16* ffh_h   = (_Float16*)A_((size_t)ROWS * FF * 2);

  auto cdiv = [](int a, int b) { return (a + b - 1) / b; };
  auto gemm = [&](const _Float16* Am, int lda, const _Float16* BT, int ldb,
                  const float* bias, float* Cf, _Float16* Ch, int ldc,
                  int M, int N, int K, int relu) {
    dim3 g(N / 16, M / 64);
    size_t smem = (size_t)16 * K * 2;   // LDS-staged B^T tile
    gemm_wmma_k<<<g, 128, smem, stream>>>(Am, lda, BT, ldb, bias, Cf, Ch, ldc, K, relu);
  };

  // ---- weight prep (f32 -> f16, transposed to [N][K]) ----
  f32_to_f16_T<<<cdiv(1024 * 512, 256), 256, 0, stream>>>(lin_in_w, h_lininT, 1024, 512);
  for (int i = 0; i < 2; ++i) {
    f32_to_f16_T<<<cdiv(512 * 512, 256), 256, 0, stream>>>(Wq + (size_t)i * 512 * 512, h_WqT[i], 512, 512);
    f32_to_f16_T<<<cdiv(512 * 512, 256), 256, 0, stream>>>(Wk + (size_t)i * 512 * 512, h_WkT[i], 512, 512);
    f32_to_f16_T<<<cdiv(512 * 512, 256), 256, 0, stream>>>(Wv + (size_t)i * 512 * 512, h_WvT[i], 512, 512);
    f32_to_f16_T<<<cdiv(512 * 2048, 256), 256, 0, stream>>>(ff1_w + (size_t)i * 512 * 2048, h_ff1T[i], 512, 2048);
    f32_to_f16_T<<<cdiv(2048 * 512, 256), 256, 0, stream>>>(ff2_w + (size_t)i * 2048 * 512, h_ff2T[i], 2048, 512);
  }
  f32_to_f16  <<<cdiv(64 * 64, 256), 256, 0, stream>>>(pos_key_emb, h_posk, 64 * 64);
  f32_to_f16_T<<<cdiv(64 * 64, 256), 256, 0, stream>>>(pos_value_emb, h_posvT, 64, 64);

  // ---- pipeline ----
  embed_k<<<ROWS, 256, 0, stream>>>(item_inputs, skill_inputs, label_inputs,
                                    item_ids, skill_ids, item_emb, skill_emb,
                                    x2e_h, query_h);
  // inputs = relu(x2e @ lin_in_w + b)
  gemm(x2e_h, 1024, h_lininT, 1024, lin_in_b, inputs_f, inputs_h, EMB,
       ROWS, EMB, 1024, 1);

  const size_t attn_smem = ((size_t)16 * SCLD + 16 * 64) * 4;  // ~71 KB
  for (int i = 0; i < 2; ++i) {
    const _Float16* qsrc = (i == 0) ? query_h : inputs_h;
    gemm(qsrc,     EMB, h_WqT[i], EMB, bq + i * EMB, nullptr, q_h, EMB, ROWS, EMB, EMB, 0);
    gemm(inputs_h, EMB, h_WkT[i], EMB, bk + i * EMB, nullptr, k_h, EMB, ROWS, EMB, EMB, 0);
    gemm(inputs_h, EMB, h_WvT[i], EMB, bv + i * EMB, nullptr, v_h, EMB, ROWS, EMB, EMB, 0);
    // qp[row][h*64+r] = q_head_h(row) . pos_key_emb[r]
    for (int hh = 0; hh < NH; ++hh)
      gemm(q_h + hh * HD, EMB, h_posk, HD, nullptr, qp_f + hh * HD, nullptr, EMB,
           ROWS, HD, HD, 0);
    attn_k<<<dim3(SEQ / 16, 8 * NH), 128, attn_smem, stream>>>(
        q_h, k_h, v_h, qp_f, h_posvT, attn_f);
    add_ln_k<<<ROWS, 256, 0, stream>>>(inputs_f, attn_f,
                                       ln_w + (size_t)(i * 2 + 0) * EMB,
                                       ln_b + (size_t)(i * 2 + 0) * EMB, inputs_h);
    gemm(inputs_h, EMB, h_ff1T[i], EMB, ff1_b + (size_t)i * FF, nullptr, ffh_h, FF,
         ROWS, FF, EMB, 1);
    gemm(ffh_h, FF, h_ff2T[i], FF, ff2_b + (size_t)i * EMB, attn_f, nullptr, EMB,
         ROWS, EMB, FF, 0);
    add_ln_k<<<ROWS, 256, 0, stream>>>(inputs_f, attn_f,
                                       ln_w + (size_t)(i * 2 + 1) * EMB,
                                       ln_b + (size_t)(i * 2 + 1) * EMB, inputs_h);
  }
  final_k<<<ROWS, 256, 0, stream>>>(inputs_f, lin_out_w, lin_out_b, (float*)d_out);
}
